// GAT_48258252537960
// MI455X (gfx1250) — matrix operand order repaired and checked by reading, not compiled, not run on hardware
//
#include <hip/hip_runtime.h>
#include <math.h>

// ---------------------------------------------------------------------------
// 2-layer GAT for MI455X (gfx1250, wave32).
//  - GEMMs via v_wmma_f32_16x16x32_bf16 on pre-packed fragment-major bf16.
//  - B fragments staged to LDS with global_load_async_to_lds_b128 (ASYNCcnt).
//  - Edge softmax + scatter-add via f32 global atomics (memory-bound path).
// ---------------------------------------------------------------------------

typedef __attribute__((ext_vector_type(16))) __bf16 v16bf;
typedef __attribute__((ext_vector_type(8)))  float  v8f;

#define NEG_SLOPE 0.2f

// ---------------- fragment packing ----------------------------------------
// A fragment layout used here (16-bit A 16x32, contiguous k order): lane {m=lane&15, h=lane>>4},
// element e -> k = h*16 + e.
// Packed: Apk[((mt*KT + kt)*32 + lane)] = v16bf fragment slice (32 bytes).
__global__ void __launch_bounds__(256)
gat_pack_a(const float* __restrict__ A, __bf16* __restrict__ Apk,
           int M, int K, int MTp) {
  const int KT = K >> 5;
  const long total = (long)MTp * KT * 32;
  const long idx = (long)blockIdx.x * blockDim.x + threadIdx.x;
  if (idx >= total) return;
  const int  lane = (int)(idx & 31);
  const long t    = idx >> 5;
  const int  kt   = (int)(t % KT);
  const int  mt   = (int)(t / KT);
  int m = mt * 16 + (lane & 15);
  if (m >= M) m = M - 1;                      // pad rows clamp (stores predicated)
  const int h = lane >> 4;
  const float* p = A + (long)m * K + kt * 32 + (h << 4);
  v16bf v;
#pragma unroll
  for (int j = 0; j < 8; ++j) {
    v[j]     = (__bf16)p[j];
    v[j + 8] = (__bf16)p[j + 8];
  }
  ((v16bf*)Apk)[idx] = v;
}

// B fragment layout (32x16): lane {n=lane&15, h=lane>>4}, element e -> k = h*16+e.
// Packed: Bpk[((kt*NT + nt)*32 + lane)] = v16bf (32 bytes).
__global__ void __launch_bounds__(256)
gat_pack_b(const float* __restrict__ B, __bf16* __restrict__ Bpk, int K, int N) {
  const int KT = K >> 5, NT = N >> 4;
  const long total = (long)KT * NT * 32;
  const long idx = (long)blockIdx.x * blockDim.x + threadIdx.x;
  if (idx >= total) return;
  const int  lane = (int)(idx & 31);
  const long t    = idx >> 5;
  const int  nt   = (int)(t % NT);
  const int  kt   = (int)(t / NT);
  const int  n    = nt * 16 + (lane & 15);
  const int  h    = lane >> 4;
  const float* p = B + (long)(kt * 32 + (h << 4)) * N + n;
  v16bf v;
#pragma unroll
  for (int e = 0; e < 16; ++e) v[e] = (__bf16)p[(long)e * N];
  ((v16bf*)Bpk)[idx] = v;
}

// ---------------- async helper ---------------------------------------------
__device__ __forceinline__ void async_b128_to_lds(unsigned lds_addr, unsigned goff,
                                                  const void* base) {
  // GVS form: global addr = 64-bit SGPR base + 32-bit VGPR offset; LDS addr from the VDST VGPR
  asm volatile("global_load_async_to_lds_b128 %0, %1, %2"
               :: "v"(lds_addr), "v"(goff), "s"(base) : "memory");
}
__device__ __forceinline__ void wait_asynccnt0() {
  asm volatile("s_wait_asynccnt 0" ::: "memory");
}

// ---------------- WMMA GEMM on packed fragments -----------------------------
// Block: 256 threads = 8 waves; tile 64(M) x 64(N).
// Wave w: mt = bx*4 + (w&3); two 16-col tiles at ntBlock + (w>>2)*2.
// Per K-step the 4 shared B fragment-sets are async-staged into LDS.
__global__ void __launch_bounds__(256)
gat_gemm_wmma_pk(const __bf16* __restrict__ Apk, const __bf16* __restrict__ Bpk,
                 float* __restrict__ C, int M, int N, int KT) {
  __shared__ __attribute__((aligned(32))) __bf16 ldsB[4 * 32 * 16];  // 4 KB

  const int tid  = threadIdx.x;
  const int wave = tid >> 5;
  const int lane = tid & 31;
  const int mt   = blockIdx.x * 4 + (wave & 3);
  const int NT   = N >> 4;
  const int ntBlock = blockIdx.y * 4;
  const int wn2  = (wave >> 2) * 2;            // 0 or 2: which staged pair

  // async-stage assignment: thread copies 16 bytes of one fragment-lane entry
  const int fsel   = tid >> 6;                 // 0..3 fragment set
  const int lanep  = (tid >> 1) & 31;
  const int half16 = tid & 1;
  const unsigned dstoff  = ((unsigned)(fsel * 32 + lanep)) * 32u + (unsigned)half16 * 16u;
  const unsigned ldsaddr = (unsigned)(size_t)(void*)ldsB + dstoff;

  const v16bf* Af    = (const v16bf*)Apk + (long)mt * KT * 32 + lane;
  const v16bf* ldsBv = (const v16bf*)ldsB;

  v8f acc0 = {};
  v8f acc1 = {};

  for (int kt = 0; kt < KT; ++kt) {
    __syncthreads();                           // previous-step consumers done
    {
      const unsigned goff =
          (((unsigned)(kt * NT + ntBlock + fsel)) * 32u + lanep) * 32u +
          (unsigned)half16 * 16u;
      async_b128_to_lds(ldsaddr, goff, Bpk);
      wait_asynccnt0();
    }
    __syncthreads();                           // staged B visible to all waves

    const v16bf a  = Af[(long)kt * 32];
    const v16bf b0 = ldsBv[(wn2 + 0) * 32 + lane];
    const v16bf b1 = ldsBv[(wn2 + 1) * 32 + lane];
    acc0 = __builtin_amdgcn_wmma_f32_16x16x32_bf16(false, a, false, b0,
                                                   (short)0, acc0, false, false);
    acc1 = __builtin_amdgcn_wmma_f32_16x16x32_bf16(false, a, false, b1,
                                                   (short)0, acc1, false, false);
  }

  // C/D layout: VGPR j -> row mt*16 + j + 8*(lane>>4), col (ntBlock+wn2)*16 + (lane&15)
  const int m0 = mt * 16;
  if (m0 < M) {                                // wave-uniform (M % 16 == 0)
    const int lm = lane & 15, lh = lane >> 4;
    float* Crow = C + (long)(m0 + (lh << 3)) * N + (ntBlock + wn2) * 16 + lm;
#pragma unroll
    for (int j = 0; j < 8; ++j) {
      Crow[(long)j * N]      = acc0[j];
      Crow[(long)j * N + 16] = acc1[j];
    }
  }
}

// ---------------- attention scores: el/er[n,h] = sum_d h[n,h,d]*a[h,d] -----
__global__ void __launch_bounds__(256)
gat_attn_scores(const float* __restrict__ h, const float* __restrict__ al,
                const float* __restrict__ ar, float* __restrict__ el,
                float* __restrict__ er, int NH, int H, int D) {
  const int pair = blockIdx.x * (blockDim.x >> 5) + (threadIdx.x >> 5);
  const int lane = threadIdx.x & 31;
  if (pair >= NH) return;
  const int n  = pair / H;
  const int hh = pair - n * H;
  const float* hp  = h  + (long)n * H * D + (long)hh * D;
  const float* alp = al + (long)hh * D;
  const float* arp = ar + (long)hh * D;
  float sl = 0.f, sr = 0.f;
  for (int d = lane; d < D; d += 32) {
    const float v = hp[d];
    sl += v * alp[d];
    sr += v * arp[d];
  }
#pragma unroll
  for (int off = 16; off > 0; off >>= 1) {
    sl += __shfl_xor(sl, off, 32);
    sr += __shfl_xor(sr, off, 32);
  }
  if (lane == 0) { el[pair] = sl; er[pair] = sr; }
}

// ---------------- helpers ---------------------------------------------------
__global__ void gat_fill(float* __restrict__ p, float v, long n) {
  long i = (long)blockIdx.x * blockDim.x + threadIdx.x;
  if (i < n) p[i] = v;
}

__device__ __forceinline__ float leaky(float v) {
  return v >= 0.f ? v : NEG_SLOPE * v;
}

__device__ __forceinline__ void atomicMaxF32(float* addr, float value) {
  if (value >= 0.f)
    atomicMax((int*)addr, __float_as_int(value));
  else
    atomicMin((unsigned int*)addr, (unsigned int)__float_as_int(value));
}

// ---------------- edge softmax passes --------------------------------------
__global__ void __launch_bounds__(256)
gat_edge_max(const int* __restrict__ src, const int* __restrict__ dst,
             const float* __restrict__ el, const float* __restrict__ er,
             float* __restrict__ emax, int EH, int H) {
  const int idx = blockIdx.x * blockDim.x + threadIdx.x;
  if (idx >= EH) return;
  const int e  = idx / H;
  const int hh = idx - e * H;
  const int s = src[e], d = dst[e];
  const float v = leaky(el[s * H + hh] + er[d * H + hh]);
  atomicMaxF32(&emax[d * H + hh], v);
}

__global__ void __launch_bounds__(256)
gat_edge_sum(const int* __restrict__ src, const int* __restrict__ dst,
             const float* __restrict__ el, const float* __restrict__ er,
             const float* __restrict__ emax, float* __restrict__ denom,
             int EH, int H) {
  const int idx = blockIdx.x * blockDim.x + threadIdx.x;
  if (idx >= EH) return;
  const int e  = idx / H;
  const int hh = idx - e * H;
  const int s = src[e], d = dst[e];
  const float v = leaky(el[s * H + hh] + er[d * H + hh]);
  atomicAdd(&denom[d * H + hh], __expf(v - emax[d * H + hh]));
}

// one wave per (edge, head); lanes stride the D=64 feature dim
__global__ void __launch_bounds__(256)
gat_edge_agg(const int* __restrict__ src, const int* __restrict__ dst,
             const float* __restrict__ el, const float* __restrict__ er,
             const float* __restrict__ emax, const float* __restrict__ denom,
             const float* __restrict__ h, float* __restrict__ out,
             int EH, int H, int D) {
  const int pair = blockIdx.x * (blockDim.x >> 5) + (threadIdx.x >> 5);
  if (pair >= EH) return;
  const int lane = threadIdx.x & 31;
  const int e  = pair / H;
  const int hh = pair - e * H;
  const int s = src[e], d = dst[e];
  const float v = leaky(el[s * H + hh] + er[d * H + hh]);
  const float w = __expf(v - emax[d * H + hh]) / denom[d * H + hh];
  const float* hp = h   + (long)s * H * D + (long)hh * D;
  float*       op = out + (long)d * H * D + (long)hh * D;
  for (int dd = lane; dd < D; dd += 32)
    atomicAdd(&op[dd], hp[dd] * w);
}

// ---------------- elementwise epilogues ------------------------------------
__global__ void gat_bias_relu(float* __restrict__ x, const float* __restrict__ b,
                              long n, int C) {
  long i = (long)blockIdx.x * blockDim.x + threadIdx.x;
  if (i < n) {
    const float v = x[i] + b[i % C];
    x[i] = v > 0.f ? v : 0.f;
  }
}

__global__ void gat_bias_add(float* __restrict__ x, const float* __restrict__ b,
                             long n, int C) {
  long i = (long)blockIdx.x * blockDim.x + threadIdx.x;
  if (i < n) x[i] += b[i % C];
}

// ---------------------------------------------------------------------------
extern "C" void kernel_launch(void* const* d_in, const int* in_sizes, int n_in,
                              void* d_out, int out_size, void* d_ws, size_t ws_size,
                              hipStream_t stream) {
  (void)n_in; (void)out_size; (void)ws_size;

  const float* feats   = (const float*)d_in[0];
  const float* W1      = (const float*)d_in[1];
  const float* attn_l1 = (const float*)d_in[2];
  const float* attn_r1 = (const float*)d_in[3];
  const float* b1      = (const float*)d_in[4];
  const float* W2      = (const float*)d_in[5];
  const float* attn_l2 = (const float*)d_in[6];
  const float* attn_r2 = (const float*)d_in[7];
  const float* b2      = (const float*)d_in[8];
  const int*   src     = (const int*)d_in[9];
  const int*   dst     = (const int*)d_in[10];
  float*       out     = (float*)d_out;

  const int IN_F = 256, H1 = 8, D = 64, F1 = H1 * D;  // 512
  const int Nn = in_sizes[0] / IN_F;                  // 50000
  const int E  = in_sizes[9];                         // 800000

  auto cdiv = [](long a, long b) { return (int)((a + b - 1) / b); };
  const int MB  = cdiv(Nn, 64);       // 782 blocks in M
  const int MTp = MB * 4;             // padded 16-row tiles (3128)
  const int KT1 = IN_F / 32, KT2 = F1 / 32;

  // workspace layout
  float* ws = (float*)d_ws;
  float* h1    = ws; ws += (long)Nn * F1;   // [N,512]
  float* out1  = ws; ws += (long)Nn * F1;   // [N,512]
  float* h2    = ws; ws += (long)Nn * D;    // [N,64]
  float* el1   = ws; ws += (long)Nn * H1;
  float* er1   = ws; ws += (long)Nn * H1;
  float* emax1 = ws; ws += (long)Nn * H1;
  float* den1  = ws; ws += (long)Nn * H1;
  float* el2   = ws; ws += Nn;
  float* er2   = ws; ws += Nn;
  float* emax2 = ws; ws += Nn;
  float* den2  = ws; ws += Nn;
  __bf16* Apk = (__bf16*)ws;                          // reused by both layers
  const long ApkElMax = (long)MTp * KT2 * 32 * 16;    // layer2 is the larger
  __bf16* Bpk1 = Apk + ApkElMax;
  __bf16* Bpk2 = Bpk1 + (long)KT1 * (F1 / 16) * 32 * 16;

  const float NEG_INF = -__builtin_huge_valf();

  // ---- Layer 1: h1 = feats @ W1 (bf16 WMMA) ----
  gat_pack_a<<<cdiv((long)MTp * KT1 * 32, 256), 256, 0, stream>>>(feats, Apk, Nn, IN_F, MTp);
  gat_pack_b<<<cdiv((long)KT1 * (F1 / 16) * 32, 256), 256, 0, stream>>>(W1, Bpk1, IN_F, F1);
  {
    dim3 g(MB, F1 / 64);
    gat_gemm_wmma_pk<<<g, 256, 0, stream>>>(Apk, Bpk1, h1, Nn, F1, KT1);
  }

  gat_attn_scores<<<cdiv((long)Nn * H1, 8), 256, 0, stream>>>(
      h1, attn_l1, attn_r1, el1, er1, Nn * H1, H1, D);

  gat_fill<<<cdiv((long)Nn * H1, 256), 256, 0, stream>>>(emax1, NEG_INF, (long)Nn * H1);
  gat_fill<<<cdiv((long)Nn * H1, 256), 256, 0, stream>>>(den1, 0.f, (long)Nn * H1);
  gat_fill<<<cdiv((long)Nn * F1, 256), 256, 0, stream>>>(out1, 0.f, (long)Nn * F1);

  gat_edge_max<<<cdiv((long)E * H1, 256), 256, 0, stream>>>(src, dst, el1, er1, emax1, E * H1, H1);
  gat_edge_sum<<<cdiv((long)E * H1, 256), 256, 0, stream>>>(src, dst, el1, er1, emax1, den1, E * H1, H1);
  gat_edge_agg<<<cdiv((long)E * H1, 8), 256, 0, stream>>>(
      src, dst, el1, er1, emax1, den1, h1, out1, E * H1, H1, D);

  gat_bias_relu<<<cdiv((long)Nn * F1, 256), 256, 0, stream>>>(out1, b1, (long)Nn * F1, F1);

  // ---- Layer 2 (H = 1): h2 = relu(out1+b1) @ W2 ----
  gat_pack_a<<<cdiv((long)MTp * KT2 * 32, 256), 256, 0, stream>>>(out1, Apk, Nn, F1, MTp);
  gat_pack_b<<<cdiv((long)KT2 * (D / 16) * 32, 256), 256, 0, stream>>>(W2, Bpk2, F1, D);
  {
    dim3 g(MB, D / 64);
    gat_gemm_wmma_pk<<<g, 256, 0, stream>>>(Apk, Bpk2, h2, Nn, D, KT2);
  }

  gat_attn_scores<<<cdiv((long)Nn, 8), 256, 0, stream>>>(
      h2, attn_l2, attn_r2, el2, er2, Nn, 1, D);

  gat_fill<<<cdiv((long)Nn, 256), 256, 0, stream>>>(emax2, NEG_INF, Nn);
  gat_fill<<<cdiv((long)Nn, 256), 256, 0, stream>>>(den2, 0.f, Nn);
  gat_fill<<<cdiv((long)Nn * D, 256), 256, 0, stream>>>(out, 0.f, (long)Nn * D);

  gat_edge_max<<<cdiv((long)E, 256), 256, 0, stream>>>(src, dst, el2, er2, emax2, E, 1);
  gat_edge_sum<<<cdiv((long)E, 256), 256, 0, stream>>>(src, dst, el2, er2, emax2, den2, E, 1);
  gat_edge_agg<<<cdiv((long)E, 8), 256, 0, stream>>>(
      src, dst, el2, er2, emax2, den2, h2, out, E, 1, D);

  gat_bias_add<<<cdiv((long)Nn * D, 256), 256, 0, stream>>>(out, b2, (long)Nn * D, D);
}
